// STBlock_4698694221990
// MI455X (gfx1250) — compile-verified, matrix operand order checked by reading, not run on hardware
//
#include <hip/hip_runtime.h>
#include <hip/hip_bf16.h>
#include <cstdint>

typedef __attribute__((ext_vector_type(16))) _Float16 v16h;
typedef __attribute__((ext_vector_type(8)))  _Float16 v8h;
typedef __attribute__((ext_vector_type(8)))  float    v8f;

#define DEV static __device__ __forceinline__

constexpr int B_ = 8, T_ = 60, N_ = 400, F_ = 64, U_ = 64, KT_ = 12;
constexpr int NP_ = 416;              // N padded to multiple of 32 (K dim)
constexpr int TP_ = 64;               // T padded to 64 (K dim for ta_out)
constexpr int NF_ = N_ * F_;          // 25600
constexpr int KC_ = KT_ * F_;         // 768 (im2col K for convs)
constexpr float NEG_ = 1e16f;

// Async global->LDS copy of 16 bytes per lane (ASYNCcnt-tracked, no VGPR data).
// LDS address = low 32 bits of the generic pointer (flat LDS aperture rule).
DEV void async_copy16(void* lds, const void* gmem) {
  uint32_t laddr = (uint32_t)(uintptr_t)lds;
  asm volatile("global_load_async_to_lds_b128 %0, %1, off"
               :: "v"(laddr), "v"(gmem) : "memory");
}
DEV void async_wait0() {
  asm volatile("s_wait_asynccnt 0" ::: "memory");
}

// ---------------- WMMA fragment helpers (V_WMMA_F32_16X16X32_F16) ----------
// A (16x32 fp16): lane L holds row M=L&15; half-index i maps to
//   K = (L>>4)*8 + i          for i in [0,8)
//   K = 16 + (L>>4)*8 + (i-8) for i in [8,16)
// => two contiguous 8-half (16B) runs per lane from a row-major tile.
// B fragments use the mirrored layout (lane holds column, same K packing),
// so loading from a TRANSPOSED B (row-major N x K) uses the same helper.
DEV v16h load_frag(const _Float16* P, int ld) {
  const int lane = threadIdx.x & 31;
  const int row  = lane & 15;
  const int kh   = lane >> 4;
  const _Float16* p = P + row * ld + kh * 8;
  v8h lo = *(const v8h*)(p);
  v8h hi = *(const v8h*)(p + 16);
  v16h r;
#pragma unroll
  for (int i = 0; i < 8; i++) { r[i] = lo[i]; r[8 + i] = hi[i]; }
  return r;
}

DEV v8f wmma32(v16h a, v16h b, v8f c) {
  return __builtin_amdgcn_wmma_f32_16x16x32_f16(false, a, false, b,
                                                (short)0, c, false, false);
}

// C/D (16x16 f32): VGPR r -> M = r + (lane>=16 ? 8 : 0), N = lane&15
DEV void store_c_f32(float* C, int ldc, v8f acc) {
  const int lane = threadIdx.x & 31, col = lane & 15, mo = (lane >> 4) * 8;
#pragma unroll
  for (int r = 0; r < 8; r++) C[(mo + r) * ldc + col] = acc[r];
}
DEV void store_c_f16(_Float16* C, int ldc, v8f acc) {
  const int lane = threadIdx.x & 31, col = lane & 15, mo = (lane >> 4) * 8;
#pragma unroll
  for (int r = 0; r < 8; r++) C[(mo + r) * ldc + col] = (_Float16)acc[r];
}

DEV float wred_max(float v) {
  for (int o = 16; o > 0; o >>= 1) v = fmaxf(v, __shfl_xor(v, o, 32));
  return v;
}
DEV float wred_sum(float v) {
  for (int o = 16; o > 0; o >>= 1) v += __shfl_xor(v, o, 32);
  return v;
}

// ---------------- prep kernels ---------------------------------------------
// xh  : (b,t,n,f)   row-major, unpadded (conv residual staging)
// xhT : (b,t,f,np)  f-major, n padded to 416 (B operand of DGC GEMMs)
__global__ void k_prep_x(const float* __restrict__ x, _Float16* __restrict__ xh,
                         _Float16* __restrict__ xhT) {
  int64_t i = (int64_t)blockIdx.x * blockDim.x + threadIdx.x;
  const int64_t tot = (int64_t)B_ * T_ * NP_ * F_;
  if (i >= tot) return;
  int f  = (int)(i & 63);
  int np = (int)((i >> 6) % NP_);
  int64_t bt = i / ((int64_t)NP_ * F_);
  float v = (np < N_) ? x[(bt * N_ + np) * F_ + f] : 0.f;
  xhT[(bt * F_ + f) * NP_ + np] = (_Float16)v;
  if (np < N_) xh[(bt * N_ + np) * F_ + f] = (_Float16)v;
}

// AWT[m][n] = A[n][m]*W1[n][m], n padded to 416 (A operand, row-major M x K)
__global__ void k_prep_awt(const float* __restrict__ A, const float* __restrict__ W1,
                           _Float16* __restrict__ AWT) {
  int i = blockIdx.x * blockDim.x + threadIdx.x;
  if (i >= N_ * NP_) return;
  int m = i / NP_, n = i % NP_;
  AWT[i] = (n < N_) ? (_Float16)(A[n * N_ + m] * W1[n * N_ + m]) : (_Float16)0.f;
}

// Transpose-convert: src (R x C) f32 row-major -> dst (C x R) fp16 row-major
__global__ void k_cvt_T(const float* __restrict__ s, _Float16* __restrict__ d,
                        int R, int C) {
  int i = blockIdx.x * blockDim.x + threadIdx.x;
  if (i >= R * C) return;
  int r = i / C, c = i % C;
  d[c * R + r] = (_Float16)s[i];
}

// zero the t = 60..63 pad columns of yT
__global__ void k_zero_yT(_Float16* __restrict__ yT) {
  int i = blockIdx.x * blockDim.x + threadIdx.x;
  if (i >= B_ * NF_ * 4) return;
  int b = i / (NF_ * 4), rem = i % (NF_ * 4);
  int nf = rem / 4, r = rem % 4;
  yT[((int64_t)b * NF_ + nf) * TP_ + (T_ + r)] = (_Float16)0.f;
}

// ---------------- fused DynamicGC (1 block per (b,t), 8 waves) --------------
// LDS: sXT 64x416 fp16 | sFeat 400x64 fp16 | per-wave 16x400 f32 stripe
__global__ __launch_bounds__(256) void k_dgc(
    const _Float16* __restrict__ xhT, const _Float16* __restrict__ AWT,
    const _Float16* __restrict__ W2T, const _Float16* __restrict__ W3T,
    const float* __restrict__ Amask, const float* __restrict__ b1,
    const float* __restrict__ b2, _Float16* __restrict__ yT) {
  extern __shared__ char smem[];
  _Float16* sXT   = (_Float16*)smem;                                  // 53248 B
  _Float16* sFeat = (_Float16*)(smem + F_ * NP_ * 2);                 // 51200 B
  float*    sDen  = (float*)(smem + F_ * NP_ * 2 + N_ * F_ * 2);      // 204800 B
  const int bt = blockIdx.x;
  const int tid = threadIdx.x, wave = tid >> 5, lane = tid & 31;

  { // async-stage x^T tile (f-major, contiguous): 13 x 16B per lane
    const char* gsrc = (const char*)(xhT + (int64_t)bt * F_ * NP_);
    char* ldst = (char*)sXT;
    for (int i = tid * 16; i < F_ * NP_ * 2; i += 256 * 16)
      async_copy16(ldst + i, gsrc + i);
    async_wait0();
  }
  __syncthreads();

  // G1: feat(400x64) = AW^T(400x416) * x(416x64)     [B from x^T in LDS]
  for (int tile = wave; tile < 100; tile += 8) {
    int mt = tile >> 2, ft = tile & 3;
    v8f acc = {};
    for (int ks = 0; ks < 13; ks++) {
      v16h a  = load_frag(AWT + mt * 16 * NP_ + ks * 32, NP_);
      v16h bf = load_frag(sXT + ft * 16 * NP_ + ks * 32, NP_);
      acc = wmma32(a, bf, acc);
    }
    store_c_f16(sFeat + mt * 16 * F_ + ft * 16, F_, acc);
  }
  __syncthreads();

  float*    myD    = sDen + wave * 16 * N_;
  _Float16* attnH  = (_Float16*)myD;                   // 16x416 fp16 (reuse)
  _Float16* nodefH = (_Float16*)((char*)myD + 16384);  // 16x64 fp16 (reuse)
  const int b = bt / T_, t = bt % T_;

  for (int mt = wave; mt < 25; mt += 8) {
    // G2: dense stripe (16x400) = feat_stripe(16x64) * W2(64x400)  [B=W2^T]
    for (int jt = 0; jt < 25; jt++) {
      v8f acc = {};
      for (int ks = 0; ks < 2; ks++) {
        v16h a  = load_frag(sFeat + mt * 16 * F_ + ks * 32, F_);
        v16h bf = load_frag(W2T + jt * 16 * F_ + ks * 32, F_);
        acc = wmma32(a, bf, acc);
      }
      store_c_f32(myD + jt * 16, N_, acc);
    }
    // bias + adjacency mask + row softmax (fp32), emit fp16 attn in place.
    // Rows processed ascending: attn row r only clobbers f32 rows < r; the
    // current row is protected by register-buffering before the stores.
    for (int r = 0; r < 16; r++) {
      int m = mt * 16 + r;
      float* row = myD + r * N_;
      float mx = -3.4e38f;
      for (int j = lane; j < N_; j += 32) {
        float v = row[j] + b1[j] - NEG_ * (1.f - Amask[m * N_ + j]);
        row[j] = v;
        mx = fmaxf(mx, v);
      }
      mx = wred_max(mx);
      float s = 0.f;
      for (int j = lane; j < N_; j += 32) {
        float e = __expf(row[j] - mx);
        row[j] = e;
        s += e;
      }
      s = wred_sum(s);
      float inv = 1.f / s;
      float vals[13];
      int c = 0;
      for (int j = lane; j < N_; j += 32) vals[c++] = row[j] * inv;
      c = 0;
      for (int j = lane; j < N_; j += 32) attnH[r * NP_ + j] = (_Float16)vals[c++];
      if (lane < 16) attnH[r * NP_ + N_ + lane] = (_Float16)0.f;  // K pad
    }
    // G3: nodef(16x64) = attn(16x416) * x(416x64)     [B from x^T in LDS]
    for (int ft = 0; ft < 4; ft++) {
      v8f acc = {};
      for (int ks = 0; ks < 13; ks++) {
        v16h a  = load_frag(attnH + ks * 32, NP_);
        v16h bf = load_frag(sXT + ft * 16 * NP_ + ks * 32, NP_);
        acc = wmma32(a, bf, acc);
      }
      store_c_f16(nodefH + ft * 16, F_, acc);
    }
    // G4: y(16x64) = nodef(16x64) * W3(64x64) + b2 -> yT[b][(m,u)][t]
    for (int ut = 0; ut < 4; ut++) {
      v8f acc = {};
      for (int ks = 0; ks < 2; ks++) {
        v16h a  = load_frag(nodefH + ks * 32, F_);
        v16h bf = load_frag(W3T + ut * 16 * F_ + ks * 32, F_);
        acc = wmma32(a, bf, acc);
      }
      const int col = lane & 15, mo = (lane >> 4) * 8;
#pragma unroll
      for (int r = 0; r < 8; r++) {
        int m = mt * 16 + mo + r, u = ut * 16 + col;
        yT[((int64_t)b * NF_ + m * U_ + u) * TP_ + t] = (_Float16)(acc[r] + b2[u]);
      }
    }
  }
}

// ---------------- temporal attention scalars (tiny, VALU) -------------------
__global__ __launch_bounds__(128) void k_ta1(
    const _Float16* __restrict__ yT, const float* __restrict__ taW1,
    const float* __restrict__ taW2, const float* __restrict__ taW3,
    float* __restrict__ lhs, float* __restrict__ rhs) {
  __shared__ float sr1[F_];
  const int bt = blockIdx.x, b = bt / T_, t = bt % T_, tid = threadIdx.x;
  const _Float16* Y = yT + (int64_t)b * NF_ * TP_;
  for (int f = tid; f < F_; f += 128) {
    float a = 0.f;
    for (int n = 0; n < N_; n++) a += (float)Y[(n * F_ + f) * TP_ + t] * taW1[n];
    sr1[f] = a;
  }
  __syncthreads();
  for (int n = tid; n < N_; n += 128) {
    float l = 0.f, rr = 0.f;
    for (int f = 0; f < F_; f++) {
      float yv = (float)Y[(n * F_ + f) * TP_ + t];
      l  += sr1[f] * taW2[f * N_ + n];
      rr += yv * taW3[f];
    }
    lhs[bt * N_ + n] = l;
    rhs[bt * N_ + n] = rr;
  }
}

__global__ __launch_bounds__(256) void k_ta2(
    const float* __restrict__ lhs, const float* __restrict__ rhs,
    const float* __restrict__ Ve, const float* __restrict__ be,
    _Float16* __restrict__ KTh) {
  __shared__ float sP[T_ * T_], sE[T_ * T_];
  const int b = blockIdx.x, tid = threadIdx.x;
  for (int idx = tid; idx < T_ * T_; idx += 256) {
    int t = idx / T_, s = idx % T_;
    const float* L = lhs + (b * T_ + t) * N_;
    const float* R = rhs + (b * T_ + s) * N_;
    float d = 0.f;
    for (int n = 0; n < N_; n++) d += L[n] * R[n];
    sP[idx] = 1.f / (1.f + __expf(-(d + be[idx])));
  }
  __syncthreads();
  for (int idx = tid; idx < T_ * T_; idx += 256) {
    int j = idx / T_, l = idx % T_;
    float e = 0.f;
    for (int k = 0; k < T_; k++) e += Ve[j * T_ + k] * sP[k * T_ + l];
    sE[idx] = e;
  }
  __syncthreads();
  for (int l = tid; l < T_; l += 256) {   // softmax over axis -2 (j)
    float mx = -3.4e38f;
    for (int j = 0; j < T_; j++) mx = fmaxf(mx, sE[j * T_ + l]);
    float s = 0.f;
    for (int j = 0; j < T_; j++) { float e = __expf(sE[j * T_ + l] - mx); sE[j * T_ + l] = e; s += e; }
    float inv = 1.f / s;
    for (int j = 0; j < T_; j++) sE[j * T_ + l] *= inv;
  }
  __syncthreads();
  // A operand for ta_out: KT[m][t] = kernel[b,t,m], padded to 64x64
  for (int idx = tid; idx < TP_ * TP_; idx += 256) {
    int m = idx >> 6, t = idx & 63;
    KTh[b * TP_ * TP_ + idx] =
        (m < T_ && t < T_) ? (_Float16)sE[t * T_ + m] : (_Float16)0.f;
  }
}

// ---------------- ta_out = KT(64x64) * Y(64 x 25600) per batch --------------
// B fragments come from yT (row = (n,u), K = t contiguous) -> 16B loads.
__global__ __launch_bounds__(256) void k_ta_out(
    const _Float16* __restrict__ KTh, const _Float16* __restrict__ yT,
    _Float16* __restrict__ tah) {
  const int b = blockIdx.x / 200, chunk = blockIdx.x % 200;
  const int wave = threadIdx.x >> 5, lane = threadIdx.x & 31;
  const _Float16* Ab = KTh + b * TP_ * TP_;
  const _Float16* Bb = yT + (int64_t)b * NF_ * TP_;
  for (int tl = wave; tl < 32; tl += 8) {
    int gt = chunk * 32 + tl;
    int nt = gt >> 2, mt = gt & 3;
    v8f acc = {};
    for (int ks = 0; ks < 2; ks++) {
      v16h a  = load_frag(Ab + mt * 16 * TP_ + ks * 32, TP_);
      v16h bf = load_frag(Bb + (int64_t)nt * 16 * TP_ + ks * 32, TP_);
      acc = wmma32(a, bf, acc);
    }
    const int col = lane & 15, mo = (lane >> 4) * 8;
#pragma unroll
    for (int r = 0; r < 8; r++) {
      int m = mt * 16 + mo + r;
      if (m < T_)
        tah[((int64_t)(b * T_) + m) * NF_ + nt * 16 + col] = (_Float16)acc[r];
    }
  }
}

// ---------------- temporal convs (im2col WMMA) + residual + softplus --------
__global__ __launch_bounds__(256) void k_conv(
    const _Float16* __restrict__ tah, const _Float16* __restrict__ xh,
    const _Float16* __restrict__ tKT, const _Float16* __restrict__ rKT,
    const float* __restrict__ tb, const float* __restrict__ rb,
    float* __restrict__ out) {
  __shared__ __align__(16) _Float16 sT[80 * 64];  // ta_out time series, pad 5/6
  __shared__ __align__(16) _Float16 sR[80 * 64];  // x time series
  const int b = blockIdx.x / N_, n = blockIdx.x % N_;
  const int tid = threadIdx.x, wave = tid >> 5, lane = tid & 31;

  // zero halo rows (p in [0,5) U [65,80)) -- disjoint from async region
  for (int i = tid; i < 20 * 64; i += 256) {
    int pr = i >> 6, c = i & 63;
    int p = (pr < 5) ? pr : (60 + pr);            // 0..4 and 65..79
    sT[p * 64 + c] = (_Float16)0.f;
    sR[p * 64 + c] = (_Float16)0.f;
  }
  // async-stage the 60 valid rows of both time series (8 x 16B per row)
  for (int idx = tid; idx < 60 * 8; idx += 256) {
    int t = idx >> 3, ch = idx & 7;
    const char* gT = (const char*)(tah + ((int64_t)(b * T_ + t)) * NF_ + n * F_) + ch * 16;
    const char* gR = (const char*)(xh + ((int64_t)(b * T_ + t) * N_ + n) * F_) + ch * 16;
    async_copy16((char*)(sT + (t + 5) * 64) + ch * 16, gT);
    async_copy16((char*)(sR + (t + 5) * 64) + ch * 16, gR);
  }
  async_wait0();
  __syncthreads();

  for (int tile = wave; tile < 16; tile += 8) {
    int mt = tile >> 2, ut = tile & 3;
    v8f acc = {};
    for (int ks = 0; ks < 24; ks++) {           // K = 12*64 = 768
      int dt = ks >> 1, c0 = (ks & 1) * 32;
      v16h a  = load_frag(sT + (mt * 16 + dt) * 64 + c0, 64);
      v16h bf = load_frag(tKT + ut * 16 * KC_ + ks * 32, KC_);
      acc = wmma32(a, bf, acc);
    }
    for (int ks = 0; ks < 24; ks++) {           // residual conv, same acc
      int dt = ks >> 1, c0 = (ks & 1) * 32;
      v16h a  = load_frag(sR + (mt * 16 + dt) * 64 + c0, 64);
      v16h bf = load_frag(rKT + ut * 16 * KC_ + ks * 32, KC_);
      acc = wmma32(a, bf, acc);
    }
    const int col = lane & 15, mo = (lane >> 4) * 8;
#pragma unroll
    for (int r = 0; r < 8; r++) {
      int m = mt * 16 + mo + r, u = ut * 16 + col;
      if (m < T_) {
        float v = acc[r] + tb[u] + rb[u];
        float sp = fmaxf(v, 0.f) + log1pf(__expf(-fabsf(v)));
        out[(((int64_t)(b * T_) + m) * N_ + n) * U_ + u] = sp;
      }
    }
  }
}

// ---------------- host launch ----------------------------------------------
extern "C" void kernel_launch(void* const* d_in, const int* in_sizes, int n_in,
                              void* d_out, int out_size, void* d_ws, size_t ws_size,
                              hipStream_t stream) {
  (void)in_sizes; (void)n_in; (void)out_size; (void)ws_size;
  const float* x      = (const float*)d_in[0];
  const float* A      = (const float*)d_in[1];
  const float* dgcW1  = (const float*)d_in[2];
  const float* dgcW2  = (const float*)d_in[3];
  const float* dgcW3  = (const float*)d_in[4];
  const float* dgcB1  = (const float*)d_in[5];
  const float* dgcB2  = (const float*)d_in[6];
  const float* taW1   = (const float*)d_in[7];
  const float* taW2   = (const float*)d_in[8];
  const float* taW3   = (const float*)d_in[9];
  const float* taVe   = (const float*)d_in[10];
  const float* taBe   = (const float*)d_in[11];
  const float* tK     = (const float*)d_in[12];
  const float* tB     = (const float*)d_in[13];
  const float* rK     = (const float*)d_in[14];
  const float* rB     = (const float*)d_in[15];
  float* out = (float*)d_out;

  char* ws = (char*)d_ws;
  size_t off = 0;
  auto carve = [&](size_t bytes) -> char* {
    off = (off + 255) & ~(size_t)255;
    char* p = ws + off;
    off += bytes;
    return p;
  };
  _Float16* xh  = (_Float16*)carve((size_t)B_ * T_ * N_ * F_ * 2);
  _Float16* xhT = (_Float16*)carve((size_t)B_ * T_ * F_ * NP_ * 2);
  _Float16* AWT = (_Float16*)carve((size_t)N_ * NP_ * 2);
  _Float16* W2T = (_Float16*)carve((size_t)N_ * F_ * 2);
  _Float16* W3T = (_Float16*)carve((size_t)U_ * F_ * 2);
  _Float16* tKT = (_Float16*)carve((size_t)U_ * KC_ * 2);
  _Float16* rKT = (_Float16*)carve((size_t)U_ * KC_ * 2);
  _Float16* yT  = (_Float16*)carve((size_t)B_ * NF_ * TP_ * 2);
  float*    lhs = (float*)carve((size_t)B_ * T_ * N_ * 4);
  float*    rhs = (float*)carve((size_t)B_ * T_ * N_ * 4);
  _Float16* KTh = (_Float16*)carve((size_t)B_ * TP_ * TP_ * 2);
  _Float16* tah = (_Float16*)carve((size_t)B_ * T_ * N_ * F_ * 2);

  const int thr = 256;
  {
    int64_t n = (int64_t)B_ * T_ * NP_ * F_;
    k_prep_x<<<(int)((n + thr - 1) / thr), thr, 0, stream>>>(x, xh, xhT);
  }
  k_prep_awt<<<(N_ * NP_ + thr - 1) / thr, thr, 0, stream>>>(A, dgcW1, AWT);
  k_cvt_T<<<(F_ * N_ + thr - 1) / thr, thr, 0, stream>>>(dgcW2, W2T, F_, N_);
  k_cvt_T<<<(F_ * U_ + thr - 1) / thr, thr, 0, stream>>>(dgcW3, W3T, F_, U_);
  k_cvt_T<<<(KC_ * U_ + thr - 1) / thr, thr, 0, stream>>>(tK, tKT, KC_, U_);
  k_cvt_T<<<(KC_ * U_ + thr - 1) / thr, thr, 0, stream>>>(rK, rKT, KC_, U_);
  k_zero_yT<<<(B_ * NF_ * 4 + thr - 1) / thr, thr, 0, stream>>>(yT);

  // fused DynamicGC: 480 workgroups, 309,248 B dynamic LDS each
  const size_t smem_dgc = (size_t)F_ * NP_ * 2 + (size_t)N_ * F_ * 2 +
                          (size_t)8 * 16 * N_ * 4;
  k_dgc<<<B_ * T_, 256, smem_dgc, stream>>>(xhT, AWT, W2T, W3T, A, dgcB1, dgcB2, yT);

  // temporal attention
  k_ta1<<<B_ * T_, 128, 0, stream>>>(yT, taW1, taW2, taW3, lhs, rhs);
  k_ta2<<<B_, 256, 0, stream>>>(lhs, rhs, taVe, taBe, KTh);
  k_ta_out<<<B_ * 200, 256, 0, stream>>>(KTh, yT, tah);

  // temporal convs + residual + softplus
  k_conv<<<B_ * N_, 256, 0, stream>>>(tah, xh, tKT, rKT, tB, rB, out);
}
